// Qwen3NextGatedDeltaNet_16758962389722
// MI455X (gfx1250) — matrix operand order repaired and checked by reading, not compiled, not run on hardware
//
#include <hip/hip_runtime.h>

// ---------------------------------------------------------------------------
// Qwen3-Next GatedDeltaNet forward for MI455X (gfx1250, wave32, WMMA + TDM)
//   B=1, S=1024, HIDDEN=2048, H_K=8, H_V=16, DK=DV=128
// Pipeline:
//   0) prep: hidden -> bf16 ; W_qkvz,W_out -> bf16 TRANSPOSED ([N][K])
//   1) qkvz = hidden @ W_qkvz          (bf16 WMMA, TDM double-buffered tiles)
//   2) ba   = hidden @ W_ba  -> beta,g
//   3) depthwise causal conv + SiLU
//   4) gated delta rule scan           (state-in-registers, 16 blocks)
//   5) gated RMSNorm  (emits bf16 directly)
//   6) out  = normed @ W_out           (bf16 WMMA, TDM double-buffered tiles)
// ---------------------------------------------------------------------------

typedef __attribute__((ext_vector_type(8)))  float         v8f;
typedef __attribute__((ext_vector_type(8)))  __bf16        bf16x8;
typedef __attribute__((ext_vector_type(16))) __bf16        bf16x16;
typedef __attribute__((ext_vector_type(4)))  unsigned int  uint32x4;
typedef __attribute__((ext_vector_type(4)))  int           int32x4;
typedef __attribute__((ext_vector_type(8)))  int           int32x8;

#if __has_builtin(__builtin_amdgcn_tensor_load_to_lds) && \
    __has_builtin(__builtin_amdgcn_s_wait_tensorcnt)
#define HAS_TDM 1
#else
#define HAS_TDM 0
#endif

__device__ __forceinline__ unsigned short f2bf(float f) {
  unsigned int u = __float_as_uint(f);
  u += 0x7FFFu + ((u >> 16) & 1u);   // round-to-nearest-even
  return (unsigned short)(u >> 16);
}

#if HAS_TDM
// ---------------------------------------------------------------------------
// Issue one TDM 2D tile load: bf16 tensor [nRows][rowLen] row-major, tile
// tile1 x tile0 at gptr (gptr already points at tile origin). LDS rows are
// padded by the TDM: +4 DWORDs after every 16 DWORDs stored, so a 64-byte
// (32 x bf16) tile row lands on an 80-byte stride == ushort[40] layout.
// Wave-level op: call from ONE wave; EXEC is ignored by tensor instructions.
// ---------------------------------------------------------------------------
__device__ __forceinline__ void tdm_load_tile(const unsigned short* gptr,
                                              unsigned lds_off,
                                              unsigned rowLen,   // tensor_dim0 (elements)
                                              unsigned nRows,    // tensor_dim1
                                              unsigned tile0,    // tile elements along row
                                              unsigned tile1) {  // tile rows
  unsigned long long ga = (unsigned long long)(size_t)gptr;
  uint32x4 g0;
  g0[0] = 1u;                                                // count=1, user D#
  g0[1] = lds_off;                                           // lds_addr (bytes)
  g0[2] = (unsigned)ga;                                      // global_addr[31:0]
  g0[3] = ((unsigned)(ga >> 32) & 0x01FFFFFFu) | (2u << 30); // ga[56:32] | type=2

  // group1: data_size=2B(code1), pad_enable, pad_interval=16DW(code3),
  //         pad_amount=4DW(code3); dims/strides per ISA bit layout.
  unsigned long long td0 = rowLen, td1 = nRows, s0 = rowLen;
  unsigned long long q0 = (1ull << 16) | (1ull << 20) | (3ull << 22) | (3ull << 25)
                        | ((td0 & 0xFFFFull) << 48);
  unsigned long long q1 = (td0 >> 16) | (td1 << 16) | ((unsigned long long)(tile0 & 0xFFFFu) << 48);
  unsigned long long q2 = (unsigned long long)(tile1 & 0xFFFFu) | (s0 << 32);
  unsigned long long q3 = (s0 >> 32) & 0xFFFFull;            // dim1_stride unused (2D tile)
  int32x8 g1;
  g1[0] = (int)(unsigned)q0; g1[1] = (int)(unsigned)(q0 >> 32);
  g1[2] = (int)(unsigned)q1; g1[3] = (int)(unsigned)(q1 >> 32);
  g1[4] = (int)(unsigned)q2; g1[5] = (int)(unsigned)(q2 >> 32);
  g1[6] = (int)(unsigned)q3; g1[7] = (int)(unsigned)(q3 >> 32);
  int32x4 g2 = {0, 0, 0, 0};
  int32x4 g3 = {0, 0, 0, 0};
#if defined(__clang_major__) && __clang_major__ >= 23
  int32x8 g4 = {0, 0, 0, 0, 0, 0, 0, 0};
  __builtin_amdgcn_tensor_load_to_lds(g0, g1, g2, g3, g4, 0);
#else
  __builtin_amdgcn_tensor_load_to_lds(g0, g1, g2, g3, 0);
#endif
}
#endif  // HAS_TDM

// ---------------------------------------------------------------------------
// GEMM: C[M,N] = A[M,K] @ Bt[N,K]^T. A,Bt bf16 row-major, C fp32.
// Block tile 64(M) x 128(N), K-step 32. 8 waves (2x4), wave owns 32x32.
// Tiles staged into LDS by the Tensor Data Mover, double-buffered:
// issue tile t+1 while computing WMMAs on tile t; sync = s_wait_tensorcnt + bar.
// Requires M%64==0, N%128==0, K%32==0.
// ---------------------------------------------------------------------------
__launch_bounds__(256)
__global__ void gemm_bf16_wmma(const unsigned short* __restrict__ A,
                               const unsigned short* __restrict__ Bt,
                               float* __restrict__ C,
                               int M, int N, int K) {
  __shared__ __attribute__((aligned(16))) unsigned short Asl[2][64][40];
  __shared__ __attribute__((aligned(16))) unsigned short Bsl[2][128][40];

  const int tid  = threadIdx.x;
  const int lane = tid & 31;
  const int wave = tid >> 5;
  const int wr   = wave >> 2;          // 0..1 (M)
  const int wc   = wave & 3;           // 0..3 (N)
  const int m0   = blockIdx.y * 64;
  const int n0   = blockIdx.x * 128;

  v8f acc[2][2] = {};

  const int l15 = lane & 15;
  const int mh  = lane >> 4;
  const int akb = mh * 8;              // A frag: K base of 8-element run
  const int bkb = mh * 8;              // B frag: K base of 8-element run
  const int nIter = K >> 5;

#if HAS_TDM
  const unsigned aoff0 = (unsigned)(size_t)&Asl[0][0][0];
  const unsigned aoff1 = (unsigned)(size_t)&Asl[1][0][0];
  const unsigned boff0 = (unsigned)(size_t)&Bsl[0][0][0];
  const unsigned boff1 = (unsigned)(size_t)&Bsl[1][0][0];
  if (wave == 0) {   // prologue: tile 0 -> buffer 0
    tdm_load_tile(A  + (size_t)m0 * K, aoff0, (unsigned)K, (unsigned)M, 32u, 64u);
    tdm_load_tile(Bt + (size_t)n0 * K, boff0, (unsigned)K, (unsigned)N, 32u, 128u);
  }
#endif

  for (int it = 0; it < nIter; ++it) {
    const int p = it & 1;
#if HAS_TDM
    if (wave == 0) __builtin_amdgcn_s_wait_tensorcnt(0);
    __syncthreads();                              // tile it visible to all waves
    if (wave == 0 && it + 1 < nIter) {            // overlap: issue tile it+1
      const int k1 = (it + 1) << 5;
      tdm_load_tile(A  + (size_t)m0 * K + k1, p ? aoff0 : aoff1,
                    (unsigned)K, (unsigned)M, 32u, 64u);
      tdm_load_tile(Bt + (size_t)n0 * K + k1, p ? boff0 : boff1,
                    (unsigned)K, (unsigned)N, 32u, 128u);
    }
#else
    {  // cooperative synchronous staging (fallback path)
      const int k0 = it << 5;
      const int r  = tid >> 2, sg = (tid & 3) * 8;
      *(uint4*)&Asl[p][r][sg] = *(const uint4*)(A + (size_t)(m0 + r) * K + k0 + sg);
      for (int t2 = tid; t2 < 512; t2 += 256) {
        const int rb = t2 >> 2, sb = (t2 & 3) * 8;
        *(uint4*)&Bsl[p][rb][sb] = *(const uint4*)(Bt + (size_t)(n0 + rb) * K + k0 + sb);
      }
      __syncthreads();
    }
#endif

    // --- fragments (ISA 7.12.2 layouts) + 4 WMMAs ---
    bf16x16 afr[2], bfr[2];
#pragma unroll
    for (int sm = 0; sm < 2; ++sm) {
      const int mm = wr * 32 + sm * 16 + l15;
      bf16x8 lo = *(const bf16x8*)&Asl[p][mm][akb];
      bf16x8 hi = *(const bf16x8*)&Asl[p][mm][16 + akb];
#pragma unroll
      for (int i = 0; i < 8; ++i) { afr[sm][i] = lo[i]; afr[sm][i + 8] = hi[i]; }
    }
#pragma unroll
    for (int sn = 0; sn < 2; ++sn) {
      const int nn = wc * 32 + sn * 16 + l15;
      bf16x8 lo = *(const bf16x8*)&Bsl[p][nn][bkb];
      bf16x8 hi = *(const bf16x8*)&Bsl[p][nn][16 + bkb];
#pragma unroll
      for (int i = 0; i < 8; ++i) { bfr[sn][i] = lo[i]; bfr[sn][i + 8] = hi[i]; }
    }
#pragma unroll
    for (int sm = 0; sm < 2; ++sm)
#pragma unroll
      for (int sn = 0; sn < 2; ++sn)
        acc[sm][sn] = __builtin_amdgcn_wmma_f32_16x16x32_bf16(
            false, afr[sm], false, bfr[sn], (short)0, acc[sm][sn], false, false);
    __syncthreads();   // done reading buffer p; safe to overwrite next round
  }

  // --- store C (C/D layout: VGPR r -> M = half*8 + r, N = lane&15) ---
  const int mh8 = mh * 8;
#pragma unroll
  for (int sm = 0; sm < 2; ++sm)
#pragma unroll
    for (int sn = 0; sn < 2; ++sn) {
      const int cm = m0 + wr * 32 + sm * 16 + mh8;
      const int cn = n0 + wc * 32 + sn * 16 + l15;
#pragma unroll
      for (int r = 0; r < 8; ++r)
        C[(size_t)(cm + r) * N + cn] = acc[sm][sn][r];
    }
}

// ---------------------------------------------------------------------------
// fp32 -> bf16 bulk convert (n % 4 == 0).
// ---------------------------------------------------------------------------
__launch_bounds__(256)
__global__ void f32_to_bf16_kernel(const float* __restrict__ in,
                                   unsigned short* __restrict__ out, int n) {
  const int i = (blockIdx.x * 256 + threadIdx.x) * 4;
  if (i >= n) return;
  float4 v = *(const float4*)(in + i);
  unsigned int lo = (unsigned)f2bf(v.x) | ((unsigned)f2bf(v.y) << 16);
  unsigned int hi = (unsigned)f2bf(v.z) | ((unsigned)f2bf(v.w) << 16);
  uint2 r; r.x = lo; r.y = hi;
  *(uint2*)(out + i) = r;
}

// ---------------------------------------------------------------------------
// Transpose + convert: in fp32 [K][N] -> out bf16 [N][K]. 32x32 LDS tiles.
// ---------------------------------------------------------------------------
__launch_bounds__(256)
__global__ void transpose_to_bf16_kernel(const float* __restrict__ in,
                                         unsigned short* __restrict__ out,
                                         int K, int N) {
  __shared__ float tile[32][33];
  const int n0 = blockIdx.x * 32;
  const int k0 = blockIdx.y * 32;
  const int tid = threadIdx.x;
  const int r = tid >> 3, c4 = (tid & 7) * 4;
  float4 v = *(const float4*)(in + (size_t)(k0 + r) * N + n0 + c4);
  tile[r][c4] = v.x; tile[r][c4 + 1] = v.y; tile[r][c4 + 2] = v.z; tile[r][c4 + 3] = v.w;
  __syncthreads();
#pragma unroll
  for (int j = 0; j < 4; ++j)
    out[(size_t)(n0 + r) * K + k0 + c4 + j] = f2bf(tile[c4 + j][r]);
}

// ---------------------------------------------------------------------------
// ba = hidden @ W_ba, then beta = sigmoid(b), g = -exp(A_log)*softplus(a+dt)
// ---------------------------------------------------------------------------
__launch_bounds__(128)
__global__ void ba_beta_g_kernel(const float* __restrict__ hidden,
                                 const float* __restrict__ Wba,
                                 const float* __restrict__ dt_bias,
                                 const float* __restrict__ A_log,
                                 float* __restrict__ beta,
                                 float* __restrict__ g) {
  __shared__ float red[128];
  const int s   = blockIdx.x;
  const int tid = threadIdx.x;
  const int col = tid & 31;
  const int seg = tid >> 5;
  const float* hrow = hidden + (size_t)s * 2048;
  float acc = 0.f;
  for (int i = seg * 512; i < seg * 512 + 512; ++i)
    acc += hrow[i] * Wba[(size_t)i * 32 + col];
  red[tid] = acc;
  __syncthreads();
  if (seg == 0) {
    float v = red[tid] + red[tid + 32] + red[tid + 64] + red[tid + 96];
    int hk = col >> 2, j = col & 3;
    if (j < 2) {
      int hv = hk * 2 + j;
      beta[s * 16 + hv] = 1.f / (1.f + expf(-v));
    } else {
      int hv = hk * 2 + (j - 2);
      float x  = v + dt_bias[hv];
      float sp = (x > 20.f) ? x : log1pf(expf(x));
      g[s * 16 + hv] = -expf(A_log[hv]) * sp;
    }
  }
}

// ---------------------------------------------------------------------------
// Depthwise causal conv (KS=4) + SiLU over re-ordered q|k|v channels.
// ---------------------------------------------------------------------------
__launch_bounds__(256)
__global__ void conv_silu_kernel(const float* __restrict__ qkvz,
                                 const float* __restrict__ convw,
                                 float* __restrict__ out) {
  const int idx = blockIdx.x * 256 + threadIdx.x;   // s*4096 + c
  const int s = idx >> 12;
  const int c = idx & 4095;
  int col;
  if (c < 1024)      { int hk = c >> 7;                     col = hk * 768 +       (c & 127); }
  else if (c < 2048) { int cc = c - 1024; int hk = cc >> 7; col = hk * 768 + 128 + (cc & 127); }
  else               { int cc = c - 2048; int hk = cc >> 8; col = hk * 768 + 256 + (cc & 255); }
  float acc = 0.f;
#pragma unroll
  for (int j = 0; j < 4; ++j) {
    int ss = s - 3 + j;
    if (ss >= 0) acc += qkvz[(size_t)ss * 6144 + col] * convw[c * 4 + j];
  }
  out[(size_t)s * 4096 + c] = acc / (1.f + expf(-acc));   // SiLU
}

// ---------------------------------------------------------------------------
// Gated delta rule scan. One block per value-head (16). 256 threads hold the
// 128x128 fp32 state in registers.
// ---------------------------------------------------------------------------
__launch_bounds__(256)
__global__ void delta_scan_kernel(const float* __restrict__ convo,
                                  const float* __restrict__ beta,
                                  const float* __restrict__ g,
                                  float* __restrict__ o) {
  const int h    = blockIdx.x;
  const int hk   = h >> 1;
  const int tid  = threadIdx.x;
  const int lane = tid & 31;
  const int v    = tid & 127;
  const int kh   = tid >> 7;
  const int kb   = kh * 64;

  __shared__ float qs[128], ks[128], vs[128], dl[128], pp[256], sc[4];

  float St[64];
#pragma unroll
  for (int i = 0; i < 64; ++i) St[i] = 0.f;

  const float scale = 0.08838834764831845f;   // 128^-0.5

  for (int s = 0; s < 1024; ++s) {
    const float* row = convo + (size_t)s * 4096;
    if (tid < 128) {
      qs[tid] = row[hk * 128 + tid];
      ks[tid] = row[1024 + hk * 128 + tid];
      vs[tid] = row[2048 + h * 128 + tid];
      if (s + 8 < 1024) {   // stream next rows toward L2/L0
        __builtin_prefetch(row + 8 * 4096 + hk * 128 + tid, 0, 1);
        __builtin_prefetch(row + 8 * 4096 + 2048 + h * 128 + tid, 0, 1);
      }
    }
    __syncthreads();

    const int wv = tid >> 5;
    if (wv == 0) {
      float a = 0.f;
#pragma unroll
      for (int j = 0; j < 4; ++j) { float x = qs[lane * 4 + j]; a += x * x; }
      a += __shfl_xor(a, 16); a += __shfl_xor(a, 8); a += __shfl_xor(a, 4);
      a += __shfl_xor(a, 2);  a += __shfl_xor(a, 1);
      if (lane == 0) sc[0] = rsqrtf(a + 1e-6f) * scale;
    } else if (wv == 1) {
      float a = 0.f;
#pragma unroll
      for (int j = 0; j < 4; ++j) { float x = ks[lane * 4 + j]; a += x * x; }
      a += __shfl_xor(a, 16); a += __shfl_xor(a, 8); a += __shfl_xor(a, 4);
      a += __shfl_xor(a, 2);  a += __shfl_xor(a, 1);
      if (lane == 0) sc[1] = rsqrtf(a + 1e-6f);
    } else if (wv == 2) {
      if (lane == 0) sc[2] = expf(g[s * 16 + h]);
      if (lane == 1) sc[3] = beta[s * 16 + h];
    }
    __syncthreads();

    const float rq = sc[0], rk = sc[1], ge = sc[2], bt = sc[3];

    float p = 0.f;
#pragma unroll
    for (int i = 0; i < 64; ++i) {
      St[i] *= ge;
      p += ks[kb + i] * rk * St[i];
    }
    pp[tid] = p;
    __syncthreads();
    if (kh == 1) dl[v] = (vs[v] - (pp[tid - 128] + p)) * bt;
    __syncthreads();

    const float d = dl[v];
    float po = 0.f;
#pragma unroll
    for (int i = 0; i < 64; ++i) {
      St[i] += ks[kb + i] * rk * d;
      po += qs[kb + i] * rq * St[i];
    }
    pp[tid] = po;
    __syncthreads();
    if (kh == 0) o[(size_t)s * 2048 + h * 128 + v] = po + pp[tid + 128];
    __syncthreads();
  }
}

// ---------------------------------------------------------------------------
// Gated RMSNorm -> bf16: x * rsqrt(mean(x^2)+eps) * w * silu(z).
// ---------------------------------------------------------------------------
__launch_bounds__(128)
__global__ void gated_rmsnorm_kernel(const float* __restrict__ o,
                                     const float* __restrict__ qkvz,
                                     const float* __restrict__ w,
                                     unsigned short* __restrict__ out) {
  __shared__ float r[4];
  const int b = blockIdx.x;
  const int s = b >> 4, h = b & 15;
  const int d = threadIdx.x;
  float x = o[(size_t)s * 2048 + h * 128 + d];
  float sq = x * x;
  sq += __shfl_xor(sq, 16); sq += __shfl_xor(sq, 8); sq += __shfl_xor(sq, 4);
  sq += __shfl_xor(sq, 2);  sq += __shfl_xor(sq, 1);
  if ((d & 31) == 0) r[d >> 5] = sq;
  __syncthreads();
  float var = (r[0] + r[1] + r[2] + r[3]) * (1.f / 128.f);
  const int hk = h >> 1, j = h & 1;
  float z = qkvz[(size_t)s * 6144 + hk * 768 + 512 + j * 128 + d];
  float res = x * rsqrtf(var + 1e-6f) * w[d] * (z / (1.f + expf(-z)));
  out[(size_t)s * 2048 + h * 128 + d] = f2bf(res);
}

// ---------------------------------------------------------------------------
extern "C" void kernel_launch(void* const* d_in, const int* in_sizes, int n_in,
                              void* d_out, int out_size, void* d_ws, size_t ws_size,
                              hipStream_t stream) {
  (void)in_sizes; (void)n_in; (void)out_size; (void)ws_size;
  const float* hidden = (const float*)d_in[0];   // [1024,2048]
  const float* Wqkvz  = (const float*)d_in[1];   // [2048,6144]
  const float* Wba    = (const float*)d_in[2];   // [2048,32]
  const float* convw  = (const float*)d_in[3];   // [4096,4]
  const float* dtb    = (const float*)d_in[4];   // [16]
  const float* Alog   = (const float*)d_in[5];   // [16]
  const float* nw     = (const float*)d_in[6];   // [128]
  const float* Wout   = (const float*)d_in[7];   // [2048,2048]
  float* out = (float*)d_out;                    // [1024,2048]

  float* ws     = (float*)d_ws;
  float* qkvz   = ws;                                   // 1024*6144 f32
  float* convo  = qkvz  + (size_t)1024 * 6144;          // 1024*4096 f32
  float* beta   = convo + (size_t)1024 * 4096;          // 1024*16
  float* gbuf   = beta  + (size_t)1024 * 16;            // 1024*16
  float* obuf   = gbuf  + (size_t)1024 * 16;            // 1024*2048 f32
  unsigned short* hid_bf  = (unsigned short*)(obuf + (size_t)1024 * 2048);
  unsigned short* norm_bf = hid_bf  + (size_t)1024 * 2048;   // [1024][2048] bf16
  unsigned short* Wt1     = norm_bf + (size_t)1024 * 2048;   // [6144][2048] bf16
  unsigned short* Wt2     = Wt1 + (size_t)6144 * 2048;       // [2048][2048] bf16

  // prep: bf16 copies (weights transposed to [N][K] so GEMM B-tiles are 2D
  // row-major blocks the TDM can stage directly)
  f32_to_bf16_kernel<<<(1024 * 2048 / 4 + 255) / 256, 256, 0, stream>>>(
      hidden, hid_bf, 1024 * 2048);
  transpose_to_bf16_kernel<<<dim3(6144 / 32, 2048 / 32), 256, 0, stream>>>(
      Wqkvz, Wt1, 2048, 6144);
  transpose_to_bf16_kernel<<<dim3(2048 / 32, 2048 / 32), 256, 0, stream>>>(
      Wout, Wt2, 2048, 2048);

  gemm_bf16_wmma<<<dim3(6144 / 128, 1024 / 64), 256, 0, stream>>>(
      hid_bf, Wt1, qkvz, 1024, 6144, 2048);
  ba_beta_g_kernel<<<1024, 128, 0, stream>>>(hidden, Wba, dtb, Alog, beta, gbuf);
  conv_silu_kernel<<<(1024 * 4096) / 256, 256, 0, stream>>>(qkvz, convw, convo);
  delta_scan_kernel<<<16, 256, 0, stream>>>(convo, beta, gbuf, obuf);
  gated_rmsnorm_kernel<<<1024 * 16, 128, 0, stream>>>(obuf, qkvz, nw, norm_bf);
  gemm_bf16_wmma<<<dim3(2048 / 128, 1024 / 64), 256, 0, stream>>>(
      norm_bf, Wt2, out, 1024, 2048, 2048);
}
